// LSTMmodel_4131758538842
// MI455X (gfx1250) — compile-verified, matrix operand order checked by reading, not run on hardware
//
#include <hip/hip_runtime.h>

// ---------------------------------------------------------------------------
// Types for CDNA5 WMMA (wave32): A/B = 16 bf16 per lane, C/D = 8 f32 per lane.
// ---------------------------------------------------------------------------
typedef __attribute__((ext_vector_type(16))) __bf16 v16bf;
typedef __attribute__((ext_vector_type(8)))  float  v8f;

__device__ __forceinline__ __bf16 cvt_bf16(float f) {
  // round-to-nearest-even via bit math
  unsigned u = __float_as_uint(f);
  u += 0x7FFFu + ((u >> 16) & 1u);
  unsigned short s = (unsigned short)(u >> 16);
  __bf16 b;
  __builtin_memcpy(&b, &s, 2);
  return b;
}

// branch-free activations (no libm divergent paths in the scan epilogue)
__device__ __forceinline__ float sigmoid_fast(float x) {
  return __builtin_amdgcn_rcpf(1.f + __expf(-x));
}
__device__ __forceinline__ float tanh_fast(float x) {
  const float xc = fminf(fmaxf(x, -10.f), 10.f);
  const float e  = __expf(2.f * xc);
  return (e - 1.f) * __builtin_amdgcn_rcpf(e + 1.f);
}

union FragBF { v16bf v; uint4 q[2]; };

// Row-major bf16 source (K contiguous per row). Per-lane layout for the
// 16-bit 16x32 A fragment (mirrored for the 32x16 B fragment):
//   lane = h*16 + m ; VGPR0-3: K = h*8+0..7 ; VGPR4-7: K = 16+h*8+0..7
__device__ __forceinline__ v16bf load_frag(const __bf16* p0, long ld, int lane) {
  int m = lane & 15, h = lane >> 4;
  const __bf16* p = p0 + (long)m * ld + h * 8;
  FragBF f;
  f.q[0] = *reinterpret_cast<const uint4*>(p);
  f.q[1] = *reinterpret_cast<const uint4*>(p + 16);
  return f.v;
}

// ---------------------------------------------------------------------------
// Batched WMMA GEMM:  out[M,N] = A[M,K](bf16) * B[N,K](bf16)^T + bias
// wave tile 64x64 (4x4 subtiles), block = 8 waves (2M x 4N) = 128x256 tile.
// MODE 0: out_bf16 = relu(acc + bias[col])   (z stage)
// MODE 1: out_f32  = acc + bias[col]         (xg stage)
// ---------------------------------------------------------------------------
template<int MODE>
__global__ void __launch_bounds__(256)
wmma_gemm(const __bf16* __restrict__ A, const __bf16* __restrict__ Bm,
          int M, int N, int K,
          const float* __restrict__ bias,
          float* __restrict__ outf, __bf16* __restrict__ outb)
{
  const int lane = threadIdx.x & 31;
  const int wave = threadIdx.x >> 5;
  const int tileM = blockIdx.y * 128 + (wave & 1) * 64;
  const int tileN = blockIdx.x * 256 + (wave >> 1) * 64;
  if (tileM >= M || tileN >= N) return;   // wave-uniform

  v8f acc[4][4] = {};

  for (int k = 0; k < K; k += 32) {
    v16bf a[4], b[4];
#pragma unroll
    for (int i = 0; i < 4; ++i)
      a[i] = load_frag(A + (long)(tileM + i * 16) * K + k, K, lane);
#pragma unroll
    for (int j = 0; j < 4; ++j)
      b[j] = load_frag(Bm + (long)(tileN + j * 16) * K + k, K, lane);
#pragma unroll
    for (int i = 0; i < 4; ++i)
#pragma unroll
      for (int j = 0; j < 4; ++j)
        acc[i][j] = __builtin_amdgcn_wmma_f32_16x16x32_bf16(
            false, a[i], false, b[j], (short)0, acc[i][j], false, false);
  }

  // C/D layout: VGPR r -> row (r + 8*h), col = lane&15
  const int m = lane & 15, h = lane >> 4;
#pragma unroll
  for (int i = 0; i < 4; ++i) {
#pragma unroll
    for (int j = 0; j < 4; ++j) {
      const int col = tileN + j * 16 + m;
      const float bv = bias[col];
#pragma unroll
      for (int r = 0; r < 8; ++r) {
        const int row = tileM + i * 16 + h * 8 + r;
        float v = acc[i][j][r] + bv;
        if (MODE == 0) {
          v = v > 0.f ? v : 0.f;
          outb[(long)row * N + col] = cvt_bf16(v);
        } else {
          outf[(long)row * N + col] = v;
        }
      }
    }
  }
}

// ---------------------------------------------------------------------------
// Fused recurrent step: per wave, a 16(batch) x 16(hidden-j) tile of ALL FOUR
// gates (accumulators at N-offsets j, j+H, j+2H, j+3H), then the LSTM
// pointwise update entirely in registers.
// K loop unrolled by 2 with two structurally distinct fragment sets
// (ping/pong) so no inter-buffer register copies are generated; the final
// prefetch address wraps with &(H-1) (no branch / no OOB read).
// Grid: 64 blocks (j-tiles of 16) x 8 waves (batch tiles of 16) = 512 waves.
// ---------------------------------------------------------------------------
__global__ void __launch_bounds__(256)
lstm_step_kernel(const __bf16* __restrict__ hb_in,   // [B,H] bf16 (prev h)
                 const __bf16* __restrict__ whhb,    // [4H,H] bf16
                 const float*  __restrict__ xg,      // [B,T,4H] f32
                 const float*  __restrict__ b_hh,    // [4H]
                 float*  __restrict__ cst,           // [B,H] f32 (in/out)
                 __bf16* __restrict__ hb_out,        // [B,H] bf16 (next h)
                 float*  __restrict__ hs_t,          // [B,H] f32 slice of hs
                 int t, int T)
{
  constexpr int H = 1024;
  constexpr long HH = (long)H * H;
  const int lane  = threadIdx.x & 31;
  const int wave  = threadIdx.x >> 5;
  const int mBase = wave * 16;               // batch rows [mBase, mBase+16)
  const int jBase = blockIdx.x * 16;         // hidden cols [jBase, jBase+16)

  const __bf16* aptr = hb_in + (long)mBase * H;
  const __bf16* bptr = whhb + (long)jBase * H;   // + g*H*H per gate

  v8f acc[4] = {};

  // prologue: fragment set 0 (k=0)
  v16bf a0 = load_frag(aptr, H, lane);
  v16bf b0[4];
#pragma unroll
  for (int g = 0; g < 4; ++g)
    b0[g] = load_frag(bptr + g * HH, H, lane);

  for (int k = 0; k < H; k += 64) {
    // fragment set 1 (k+32) — issue loads, then consume set 0
    v16bf a1 = load_frag(aptr + k + 32, H, lane);
    v16bf b1[4];
#pragma unroll
    for (int g = 0; g < 4; ++g)
      b1[g] = load_frag(bptr + g * HH + k + 32, H, lane);
#pragma unroll
    for (int g = 0; g < 4; ++g)
      acc[g] = __builtin_amdgcn_wmma_f32_16x16x32_bf16(
          false, a0, false, b0[g], (short)0, acc[g], false, false);

    // fragment set 0 (k+64, wraps to 0 on last iter) — then consume set 1
    const int kn = (k + 64) & (H - 1);
    a0 = load_frag(aptr + kn, H, lane);
#pragma unroll
    for (int g = 0; g < 4; ++g)
      b0[g] = load_frag(bptr + g * HH + kn, H, lane);
#pragma unroll
    for (int g = 0; g < 4; ++g)
      acc[g] = __builtin_amdgcn_wmma_f32_16x16x32_bf16(
          false, a1, false, b1[g], (short)0, acc[g], false, false);
  }

  // epilogue: lane holds col j = jBase + (lane&15), rows mBase + 8*h + r
  const int m = lane & 15, h = lane >> 4;
  const int j = jBase + m;
  float bias_g[4];
#pragma unroll
  for (int g = 0; g < 4; ++g) bias_g[g] = b_hh[g * H + j];

#pragma unroll
  for (int r = 0; r < 8; ++r) {
    const int b = mBase + h * 8 + r;
    const float* xg_row = xg + ((long)b * T + t) * (4 * H);
    const float ig = acc[0][r] + bias_g[0] + xg_row[0 * H + j];
    const float fg = acc[1][r] + bias_g[1] + xg_row[1 * H + j];
    const float gg = acc[2][r] + bias_g[2] + xg_row[2 * H + j];
    const float og = acc[3][r] + bias_g[3] + xg_row[3 * H + j];
    const long idx = (long)b * H + j;
    const float cn = sigmoid_fast(fg) * cst[idx] + sigmoid_fast(ig) * tanh_fast(gg);
    cst[idx] = cn;
    const float hn = sigmoid_fast(og) * tanh_fast(cn);
    hs_t[idx] = hn;
    hb_out[idx] = cvt_bf16(hn);
  }
}

// ---------------------------------------------------------------------------
__global__ void f32_to_bf16_kernel(const float* __restrict__ in,
                                   __bf16* __restrict__ out, long n) {
  long i = (long)blockIdx.x * blockDim.x + threadIdx.x;
  const long stride = (long)gridDim.x * blockDim.x;
  for (; i < n; i += stride) out[i] = cvt_bf16(in[i]);
}

__global__ void init_state_kernel(float* __restrict__ c,
                                  __bf16* __restrict__ hb, int n) {
  int i = blockIdx.x * blockDim.x + threadIdx.x;
  if (i < n) { c[i] = 0.f; hb[i] = cvt_bf16(0.f); }
}

// scores[b*T+t, c] = hs[t,b,:] . Wc[c,:] + bc[c]   (C=22, fp32 exact)
__global__ void classifier_kernel(const float* __restrict__ hs,
                                  const float* __restrict__ Wc,
                                  const float* __restrict__ bc,
                                  float* __restrict__ out,
                                  int T, int Bb, int Hh, int C) {
  const int row = blockIdx.x;             // b*T + t
  const int b = row / T, t = row % T;
  const float* hr = hs + ((long)t * Bb + b) * Hh;
  const int c = threadIdx.x;
  if (c >= C) return;
  const float* wr = Wc + (long)c * Hh;
  float s = 0.f;
  for (int k = 0; k < Hh; k += 4) {
    const float4 hv = *reinterpret_cast<const float4*>(hr + k);
    const float4 wv = *reinterpret_cast<const float4*>(wr + k);
    s += hv.x * wv.x + hv.y * wv.y + hv.z * wv.z + hv.w * wv.w;
  }
  out[(long)row * C + c] = s + bc[c];
}

// ---------------------------------------------------------------------------
extern "C" void kernel_launch(void* const* d_in, const int* in_sizes, int n_in,
                              void* d_out, int out_size, void* d_ws, size_t ws_size,
                              hipStream_t stream) {
  const float* x    = (const float*)d_in[0];
  const float* W1   = (const float*)d_in[1];
  const float* b1   = (const float*)d_in[2];
  const float* W_ih = (const float*)d_in[3];
  const float* b_ih = (const float*)d_in[4];
  const float* W_hh = (const float*)d_in[5];
  const float* b_hh = (const float*)d_in[6];
  const float* Wc   = (const float*)d_in[7];
  const float* bc   = (const float*)d_in[8];
  float* out = (float*)d_out;

  constexpr int  Bb = 128, T = 64, F = 2048, N = 1024, H = 1024, C = 22;
  constexpr long MROWS = (long)Bb * T;          // 8192

  // workspace carve (256B aligned): ~240 MB total
  char* p = (char*)d_ws;
  auto take = [&](long bytes) { char* r = p; p += (bytes + 255) & ~255L; return r; };
  __bf16* xb    = (__bf16*)take(MROWS * F * 2);          // 32 MB
  __bf16* w1b   = (__bf16*)take((long)N * F * 2);        //  4 MB
  __bf16* wihb  = (__bf16*)take((long)4 * H * N * 2);    //  8 MB
  __bf16* whhb  = (__bf16*)take((long)4 * H * H * 2);    //  8 MB
  __bf16* zb    = (__bf16*)take(MROWS * N * 2);          // 16 MB
  float*  xg    = (float*) take(MROWS * 4 * H * 4);      // 128 MB
  float*  cst   = (float*) take((long)Bb * H * 4);
  __bf16* hbA   = (__bf16*)take((long)Bb * H * 2);       // ping
  __bf16* hbB   = (__bf16*)take((long)Bb * H * 2);       // pong
  float*  hs    = (float*) take((long)T * Bb * H * 4);   // 32 MB

  // fp32 -> bf16 operand copies
  f32_to_bf16_kernel<<<2048, 256, 0, stream>>>(x, xb, MROWS * F);
  f32_to_bf16_kernel<<<1024, 256, 0, stream>>>(W1, w1b, (long)N * F);
  f32_to_bf16_kernel<<<1024, 256, 0, stream>>>(W_ih, wihb, (long)4 * H * N);
  f32_to_bf16_kernel<<<1024, 256, 0, stream>>>(W_hh, whhb, (long)4 * H * H);

  // z = relu(x W1^T + b1)  -> bf16 [8192,1024]
  dim3 gz(N / 256, MROWS / 128);
  wmma_gemm<0><<<gz, 256, 0, stream>>>(xb, w1b, (int)MROWS, N, F,
                                       b1, nullptr, zb);

  // xg = z W_ih^T + b_ih   -> f32 [8192,4096]
  dim3 gx(4 * H / 256, MROWS / 128);
  wmma_gemm<1><<<gx, 256, 0, stream>>>(zb, wihb, (int)MROWS, 4 * H, N,
                                       b_ih, xg, nullptr);

  init_state_kernel<<<(Bb * H + 255) / 256, 256, 0, stream>>>(cst, hbA, Bb * H);

  // recurrent scan: one fused WMMA+pointwise kernel per step (ping-pong h)
  for (int t = 0; t < T; ++t) {
    const __bf16* hin  = (t & 1) ? hbB : hbA;
    __bf16*       hout = (t & 1) ? hbA : hbB;
    lstm_step_kernel<<<H / 16, 256, 0, stream>>>(hin, whhb, xg, b_hh, cst,
                                                 hout, hs + (long)t * Bb * H,
                                                 t, T);
  }

  // scores = hs Wc^T + bc  -> [B*T, 22]
  classifier_kernel<<<(int)MROWS, 32, 0, stream>>>(hs, Wc, bc, out, T, Bb, H, C);
}
